// SecondOrderDynamics_54494545051765
// MI455X (gfx1250) — matrix operand order repaired and checked by reading, not compiled, not run on hardware
//
#include <hip/hip_runtime.h>

typedef __attribute__((ext_vector_type(16))) __bf16 v16bf;
typedef __attribute__((ext_vector_type(8)))  __bf16 v8bf;
typedef __attribute__((ext_vector_type(8)))  float  v8f;

#define BATCH 2048
#define NDZ   (BATCH * 256)   // elements in a [B,256] state tensor

__device__ __forceinline__ float sigm_f(float x){ return 1.0f/(1.0f + __expf(-x)); }
__device__ __forceinline__ float silu_f(float x){ return x * sigm_f(x); }
__device__ __forceinline__ float silu_d(float x){ float s = sigm_f(x); return s*(1.0f + x*(1.0f - s)); }

__device__ __forceinline__ v16bf cat8(v8bf lo, v8bf hi){
  return __builtin_shufflevector(lo, hi, 0,1,2,3,4,5,6,7,8,9,10,11,12,13,14,15);
}

// ---------------------------------------------------------------------------
// Generic bf16 WMMA GEMM:  out = epilogue( A[M,K] @ Bt^T + bias ),
// Bt stored as [N,K] row-major (i.e. W^T) so fragments are contiguous loads.
// Block = 128 threads (4 waves), tile = 128x64; wave w owns rows
// [w*32, w*32+32) as two 16-row fragments sharing four B fragments
// (8 WMMAs per 12 b128-loads per K-step -> ~21 FLOP/B of cache traffic).
// Epilogue modes:
//  0: outF  = x                     5: outF  = -x
//  1: outH  = bf16(silu(x)), spOut = silu'(x)
//  2: outH  = bf16(silu(x))         6: outF += x
//  3: outH  = bf16(auxVec[col]*silu'(x))
//  4: outH  = bf16(x * auxMat[row,col])   7: outF += scale[0]*x
//  8: outF  = silu(x)               9: outH  = bf16(x)
// ---------------------------------------------------------------------------
__global__ __launch_bounds__(128) void gemm_bf16_wmma(
    const __bf16* __restrict__ A, int lda,
    const __bf16* __restrict__ Bt,
    const float*  __restrict__ bias,
    int K, int mode,
    float* __restrict__ outF, int ldf,
    __bf16* __restrict__ outH, int ldh,
    float* __restrict__ spOut,
    const float* __restrict__ auxVec,
    const float* __restrict__ auxMat, int ldaux,
    const float* __restrict__ scalePtr)
{
  const int lane = threadIdx.x & 31;
  const int wave = threadIdx.x >> 5;
  const int half = lane >> 4;
  const int l15  = lane & 15;
  const int rowA = blockIdx.y * 128 + wave * 32 + l15;  // A fragment 0 row (M)
  const int colB = blockIdx.x * 64 + l15;               // B fragment base col (N)

  v8f c0 = {}, c1 = {}, c2 = {}, c3 = {};   // rows [w*32, w*32+16)
  v8f c4 = {}, c5 = {}, c6 = {}, c7 = {};   // rows [w*32+16, w*32+32)
  const __bf16* pArow0 = A + (size_t)rowA * lda;
  const __bf16* pArow1 = pArow0 + (size_t)16 * lda;
  const __bf16* pBrow  = Bt + (size_t)colB * K;

  for (int kk = 0; kk < K; kk += 32) {
    // A in contiguous k order, not the ISA table order: lanes 0-15 hold K[0..15], lanes 16-31 K[16..31] (matches B)
    v16bf a0 = *(const v16bf*)(pArow0 + kk + half * 16);
    v16bf a1 = *(const v16bf*)(pArow1 + kk + half * 16);
    // B in the same contiguous k order: lanes 0-15 K[0..15], lanes 16-31 K[16..31], col = lane&15
    const __bf16* pb = pBrow + kk + half * 16;
    v16bf b0 = *(const v16bf*)(pb);
    v16bf b1 = *(const v16bf*)(pb + (size_t)16 * K);
    v16bf b2 = *(const v16bf*)(pb + (size_t)32 * K);
    v16bf b3 = *(const v16bf*)(pb + (size_t)48 * K);
    c0 = __builtin_amdgcn_wmma_f32_16x16x32_bf16(false, a0, false, b0, (short)0, c0, false, false);
    c4 = __builtin_amdgcn_wmma_f32_16x16x32_bf16(false, a1, false, b0, (short)0, c4, false, false);
    c1 = __builtin_amdgcn_wmma_f32_16x16x32_bf16(false, a0, false, b1, (short)0, c1, false, false);
    c5 = __builtin_amdgcn_wmma_f32_16x16x32_bf16(false, a1, false, b1, (short)0, c5, false, false);
    c2 = __builtin_amdgcn_wmma_f32_16x16x32_bf16(false, a0, false, b2, (short)0, c2, false, false);
    c6 = __builtin_amdgcn_wmma_f32_16x16x32_bf16(false, a1, false, b2, (short)0, c6, false, false);
    c3 = __builtin_amdgcn_wmma_f32_16x16x32_bf16(false, a0, false, b3, (short)0, c3, false, false);
    c7 = __builtin_amdgcn_wmma_f32_16x16x32_bf16(false, a1, false, b3, (short)0, c7, false, false);
  }

  v8f acc[2][4] = {{c0, c1, c2, c3}, {c4, c5, c6, c7}};
  #pragma unroll
  for (int rf = 0; rf < 2; ++rf) {
    const int rowO = blockIdx.y * 128 + wave * 32 + rf * 16 + half * 8;  // C/D layout rows
    #pragma unroll
    for (int nb = 0; nb < 4; ++nb) {
      const int col = blockIdx.x * 64 + nb * 16 + l15;
      const float bcol = bias ? bias[col] : 0.0f;
      #pragma unroll
      for (int r = 0; r < 8; ++r) {
        const int rr = rowO + r;
        float x = acc[rf][nb][r] + bcol;
        switch (mode) {
          case 0: outF[(size_t)rr * ldf + col] = x; break;
          case 1: outH[(size_t)rr * ldh + col] = (__bf16)silu_f(x);
                  spOut[(size_t)rr * ldh + col] = silu_d(x); break;
          case 2: outH[(size_t)rr * ldh + col] = (__bf16)silu_f(x); break;
          case 3: outH[(size_t)rr * ldh + col] = (__bf16)(auxVec[col] * silu_d(x)); break;
          case 4: outH[(size_t)rr * ldh + col] = (__bf16)(x * auxMat[(size_t)rr * ldaux + col]); break;
          case 5: outF[(size_t)rr * ldf + col] = -x; break;
          case 6: outF[(size_t)rr * ldf + col] += x; break;
          case 7: outF[(size_t)rr * ldf + col] += scalePtr[0] * x; break;
          case 8: outF[(size_t)rr * ldf + col] = silu_f(x); break;
          case 9: outH[(size_t)rr * ldh + col] = (__bf16)x; break;
        }
      }
    }
  }
}

// ------------------------- elementwise helpers -----------------------------
__global__ void k_cvt_trans(const float* __restrict__ s, __bf16* __restrict__ d, int R, int C){
  int i = blockIdx.x * blockDim.x + threadIdx.x;
  if (i >= R * C) return;
  int r = i / C, c = i - r * C;
  d[(size_t)c * R + r] = (__bf16)s[i];          // dst is [C,R]
}
__global__ void k_cvt_copy(const float* __restrict__ s, __bf16* __restrict__ d, int n){
  int i = blockIdx.x * blockDim.x + threadIdx.x;
  if (i < n) d[i] = (__bf16)s[i];
}
__global__ void k_copy(const float* __restrict__ s, float* __restrict__ d, int n){
  int i = blockIdx.x * blockDim.x + threadIdx.x;
  if (i < n) d[i] = s[i];
}
__global__ void k_build_zv(const float* __restrict__ z, const float* __restrict__ v,
                           __bf16* __restrict__ zv, int n){
  int i = blockIdx.x * blockDim.x + threadIdx.x;
  if (i >= n) return;
  int b = i >> 9, c = i & 511;
  float x = (c < 256) ? z[b * 256 + c] : v[b * 256 + (c - 256)];
  zv[i] = (__bf16)x;
}
__global__ void k_drive_bias(const float* __restrict__ db1, const float* __restrict__ dW1,
                             float t, float* __restrict__ o){
  int i = blockIdx.x * blockDim.x + threadIdx.x;
  if (i < 512) o[i] = db1[i] + t * dW1[256 * 512 + i];   // fold t-column into bias
}
__global__ void k_reduce_alpha(const float* __restrict__ ah, const float* __restrict__ cW2,
                               const float* __restrict__ cb2, float* __restrict__ alpha){
  int bg = blockIdx.x; int g = bg & 3; int b = bg >> 2;
  const float* row = ah + (size_t)b * 2048 + g * 512;
  const float* w   = cW2 + g * 512;
  float s = 0.f;
  for (int h = threadIdx.x; h < 512; h += 128) s += row[h] * w[h];
  __shared__ float red[128];
  red[threadIdx.x] = s; __syncthreads();
  for (int off = 64; off > 0; off >>= 1) {
    if (threadIdx.x < off) red[threadIdx.x] += red[threadIdx.x + off];
    __syncthreads();
  }
  if (threadIdx.x == 0) alpha[bg] = red[0] + cb2[g];
}
__global__ void k_combine(const float* __restrict__ Facc, const float* __restrict__ vstage,
                          const float* __restrict__ alpha, const float* __restrict__ GF,
                          const float* __restrict__ damping, const float* __restrict__ pw,
                          const float* __restrict__ sw, float* __restrict__ aF, int n){
  int i = blockIdx.x * blockDim.x + threadIdx.x;
  if (i >= n) return;
  int b = i >> 8, d = i & 255;
  float sp = log1pf(__expf(damping[0]));           // softplus
  float wp = 1.f / (1.f + __expf(-pw[0]));
  float ws = 1.f / (1.f + __expf(-sw[0]));
  float fs = 0.f;
  #pragma unroll
  for (int g = 0; g < 4; ++g) fs += alpha[b * 4 + g] * GF[(size_t)b * 1024 + g * 256 + d];
  aF[i] = wp * (Facc[i] - sp * vstage[i]) + ws * fs;
}
__global__ void k_lincomb(float* __restrict__ o, const float* __restrict__ x,
                          const float* __restrict__ y, float s, int n){
  int i = blockIdx.x * blockDim.x + threadIdx.x;
  if (i < n) o[i] = x[i] + s * y[i];
}
__global__ void k_accum(float* __restrict__ o, const float* __restrict__ y, float s, int n){
  int i = blockIdx.x * blockDim.x + threadIdx.x;
  if (i < n) o[i] += s * y[i];
}
__global__ void k_concat(const float* __restrict__ z, const float* __restrict__ v,
                         float* __restrict__ o, int n){
  int i = blockIdx.x * blockDim.x + threadIdx.x;
  if (i >= n) return;
  int b = i >> 9, c = i & 511;
  o[i] = (c < 256) ? z[b * 256 + c] : v[b * 256 + (c - 256)];
}

static inline dim3 ewg(int n){ return dim3((n + 255) / 256); }

static void gemm(hipStream_t st, const __bf16* A, int lda, const __bf16* Bt, const float* bias,
                 int M, int N, int K, int mode,
                 float* outF, int ldf, __bf16* outH, int ldh,
                 float* spOut, const float* auxVec, const float* auxMat, int ldaux,
                 const float* scalePtr)
{
  dim3 g(N / 64, M / 128);
  gemm_bf16_wmma<<<g, 128, 0, st>>>(A, lda, Bt, bias, K, mode, outF, ldf, outH, ldh,
                                    spOut, auxVec, auxMat, ldaux, scalePtr);
}

extern "C" void kernel_launch(void* const* d_in, const int* in_sizes, int n_in,
                              void* d_out, int out_size, void* d_ws, size_t ws_size,
                              hipStream_t stream) {
  (void)in_sizes; (void)n_in; (void)out_size; (void)ws_size;
  const float* z_in    = (const float*)d_in[0];
  const float* v_in    = (const float*)d_in[1];
  const float* pW1     = (const float*)d_in[2];
  const float* pb1     = (const float*)d_in[3];
  const float* pW2     = (const float*)d_in[4];
  const float* pb2     = (const float*)d_in[5];
  const float* pW3     = (const float*)d_in[6];
  /* pb3 = d_in[7] unused: constant bias vanishes under grad */
  const float* damping = (const float*)d_in[8];
  const float* dW1     = (const float*)d_in[9];
  const float* db1     = (const float*)d_in[10];
  const float* dW2     = (const float*)d_in[11];
  const float* db2     = (const float*)d_in[12];
  const float* nW1     = (const float*)d_in[13];
  const float* nb1     = (const float*)d_in[14];
  const float* nW2     = (const float*)d_in[15];
  const float* nb2     = (const float*)d_in[16];
  const float* nscale  = (const float*)d_in[17];
  const float* G       = (const float*)d_in[18];
  const float* cW1     = (const float*)d_in[19];
  const float* cb1     = (const float*)d_in[20];
  const float* cW2     = (const float*)d_in[21];
  const float* cb2     = (const float*)d_in[22];
  const float* eW1     = (const float*)d_in[23];
  const float* eb1     = (const float*)d_in[24];
  const float* eW2     = (const float*)d_in[25];
  const float* eb2     = (const float*)d_in[26];
  const float* pweight = (const float*)d_in[27];
  const float* sweight = (const float*)d_in[28];

  // ---------------- workspace bump allocator ----------------
  char* base = (char*)d_ws; size_t off = 0;
  auto alloc = [&](size_t bytes){ char* r = base + off; off += (bytes + 255) & ~(size_t)255; return (void*)r; };

  __bf16* w_pW1t = (__bf16*)alloc((size_t)512 * 256 * 2);   // [N=512,K=256] = pW1^T
  __bf16* w_pW2t = (__bf16*)alloc((size_t)512 * 512 * 2);   // pW2^T
  __bf16* w_pW2d = (__bf16*)alloc((size_t)512 * 512 * 2);   // pW2 (Bt for W^T gemm)
  __bf16* w_pW1d = (__bf16*)alloc((size_t)256 * 512 * 2);   // pW1 (Bt for W^T gemm)
  __bf16* w_dW1z = (__bf16*)alloc((size_t)512 * 256 * 2);   // dW1[0:256,:]^T
  __bf16* w_dW2  = (__bf16*)alloc((size_t)256 * 512 * 2);
  __bf16* w_nW1  = (__bf16*)alloc((size_t)512 * 512 * 2);
  __bf16* w_nW2  = (__bf16*)alloc((size_t)256 * 512 * 2);
  __bf16* w_c1   = (__bf16*)alloc((size_t)2048 * 512 * 2);  // [g*512+h, k]
  __bf16* w_eW1  = (__bf16*)alloc((size_t)512 * 256 * 2);
  __bf16* w_eW2  = (__bf16*)alloc((size_t)256 * 512 * 2);
  __bf16* w_G    = (__bf16*)alloc((size_t)1024 * 256 * 2);  // [g*256+i, j] = G flat
  __bf16* zv_bf  = (__bf16*)alloc((size_t)BATCH * 512 * 2); // [z|v] rows; z view = lda 512
  __bf16* hA     = (__bf16*)alloc((size_t)BATCH * 512 * 2);
  __bf16* hB     = (__bf16*)alloc((size_t)BATCH * 512 * 2);
  __bf16* Fb_bf  = (__bf16*)alloc((size_t)BATCH * 256 * 2);
  float*  sp1    = (float*)alloc((size_t)BATCH * 512 * 4);
  float*  Facc   = (float*)alloc((size_t)BATCH * 256 * 4);
  float*  ah     = (float*)alloc((size_t)BATCH * 2048 * 4);
  float*  GF     = (float*)alloc((size_t)BATCH * 1024 * 4);
  float*  alpha  = (float*)alloc((size_t)BATCH * 4 * 4);
  float*  db1_t  = (float*)alloc((size_t)512 * 4);
  float*  zcur   = (float*)alloc((size_t)NDZ * 4);
  float*  vcur   = (float*)alloc((size_t)NDZ * 4);
  float*  zacc   = (float*)alloc((size_t)NDZ * 4);
  float*  vacc   = (float*)alloc((size_t)NDZ * 4);
  float*  zs     = (float*)alloc((size_t)NDZ * 4);
  float*  vs     = (float*)alloc((size_t)NDZ * 4);
  float*  aF     = (float*)alloc((size_t)NDZ * 4);

  // ---------------- one-time bf16 weight prep ----------------
  auto cvtT = [&](const float* s, __bf16* d, int R, int C){
    int n = R * C; k_cvt_trans<<<ewg(n), 256, 0, stream>>>(s, d, R, C);
  };
  auto cvtC = [&](const float* s, __bf16* d, int n){
    k_cvt_copy<<<ewg(n), 256, 0, stream>>>(s, d, n);
  };
  cvtT(pW1, w_pW1t, 256, 512);
  cvtT(pW2, w_pW2t, 512, 512);
  cvtC(pW2, w_pW2d, 512 * 512);
  cvtC(pW1, w_pW1d, 256 * 512);
  cvtT(dW1, w_dW1z, 256, 512);                 // only first 256 rows of [257,512]
  cvtT(dW2, w_dW2, 512, 256);
  cvtT(nW1, w_nW1, 512, 512);
  cvtT(nW2, w_nW2, 512, 256);
  for (int g = 0; g < 4; ++g) cvtT(cW1 + (size_t)g * 512 * 512, w_c1 + (size_t)g * 512 * 512, 512, 512);
  cvtT(eW1, w_eW1, 256, 512);
  cvtT(eW2, w_eW2, 512, 256);
  cvtC(G, w_G, 1024 * 256);
  k_copy<<<ewg(NDZ), 256, 0, stream>>>(z_in, zcur, NDZ);
  k_copy<<<ewg(NDZ), 256, 0, stream>>>(v_in, vcur, NDZ);

  // ---------------- one force evaluation ----------------
  auto force = [&](const float* zp, const float* vp, float t){
    k_build_zv<<<ewg(BATCH * 512), 256, 0, stream>>>(zp, vp, zv_bf, BATCH * 512);
    k_drive_bias<<<2, 256, 0, stream>>>(db1, dW1, t, db1_t);
    // potential fwd: a1 = z@pW1+pb1 -> h1(bf16)=silu, sp1=silu'
    gemm(stream, zv_bf, 512, w_pW1t, pb1, BATCH, 512, 256, 1, nullptr, 0, hA, 512, sp1, nullptr, nullptr, 0, nullptr);
    // g_a2 = silu'(h1@pW2+pb2) * pW3
    gemm(stream, hA, 512, w_pW2t, pb2, BATCH, 512, 512, 3, nullptr, 0, hB, 512, nullptr, pW3, nullptr, 0, nullptr);
    // g_a1 = (g_a2 @ pW2^T) * sp1
    gemm(stream, hB, 512, w_pW2d, nullptr, BATCH, 512, 512, 4, nullptr, 0, hA, 512, nullptr, nullptr, sp1, 512, nullptr);
    // F_cons: Facc = -(g_a1 @ pW1^T)
    gemm(stream, hA, 512, w_pW1d, nullptr, BATCH, 256, 512, 5, Facc, 256, nullptr, 0, nullptr, nullptr, nullptr, 0, nullptr);
    // drive: hd = silu(z@dW1z + (db1 + t*dW1_row))
    gemm(stream, zv_bf, 512, w_dW1z, db1_t, BATCH, 512, 256, 2, nullptr, 0, hB, 512, nullptr, nullptr, nullptr, 0, nullptr);
    gemm(stream, hB, 512, w_dW2, db2, BATCH, 256, 512, 6, Facc, 256, nullptr, 0, nullptr, nullptr, nullptr, 0, nullptr);
    // nonlinear: hn = silu(zv@nW1+nb1); Facc += scale*(hn@nW2+nb2)
    gemm(stream, zv_bf, 512, w_nW1, nb1, BATCH, 512, 512, 2, nullptr, 0, hB, 512, nullptr, nullptr, nullptr, 0, nullptr);
    gemm(stream, hB, 512, w_nW2, nb2, BATCH, 256, 512, 7, Facc, 256, nullptr, 0, nullptr, nullptr, nullptr, 0, nscale);
    // coeff nets: ah = silu(zv@cW1cat + cb1) ; alpha = <ah, cW2> + cb2
    gemm(stream, zv_bf, 512, w_c1, cb1, BATCH, 2048, 512, 8, ah, 2048, nullptr, 0, nullptr, nullptr, nullptr, 0, nullptr);
    k_reduce_alpha<<<BATCH * 4, 128, 0, stream>>>(ah, cW2, cb2, alpha);
    // equivariant: Fbase = silu(z@eW1+eb1)@eW2+eb2 (bf16)
    gemm(stream, zv_bf, 512, w_eW1, eb1, BATCH, 512, 256, 2, nullptr, 0, hB, 512, nullptr, nullptr, nullptr, 0, nullptr);
    gemm(stream, hB, 512, w_eW2, eb2, BATCH, 256, 512, 9, nullptr, 0, Fb_bf, 256, nullptr, nullptr, nullptr, 0, nullptr);
    // GF[b, g*256+i] = sum_j G[g,i,j]*Fbase[b,j]
    gemm(stream, Fb_bf, 256, w_G, nullptr, BATCH, 1024, 256, 0, GF, 1024, nullptr, 0, nullptr, nullptr, nullptr, 0, nullptr);
    k_combine<<<ewg(NDZ), 256, 0, stream>>>(Facc, vp, alpha, GF, damping, pweight, sweight, aF, NDZ);
  };

  // ---------------- RK4 integration ----------------
  const float DT = 0.1f;
  float* zp = zcur; float* vp = vcur; float* za = zacc; float* va = vacc;
  for (int step = 0; step < 10; ++step) {
    float t = step * DT;
    // stage 1 (k1v = v, k1a = aF)
    force(zp, vp, t);
    k_lincomb<<<ewg(NDZ), 256, 0, stream>>>(za, zp, vp, DT / 6.f, NDZ);
    k_lincomb<<<ewg(NDZ), 256, 0, stream>>>(va, vp, aF, DT / 6.f, NDZ);
    k_lincomb<<<ewg(NDZ), 256, 0, stream>>>(zs, zp, vp, DT / 2.f, NDZ);
    k_lincomb<<<ewg(NDZ), 256, 0, stream>>>(vs, vp, aF, DT / 2.f, NDZ);
    // stage 2 (k2v = vs, k2a = aF)
    force(zs, vs, t + DT / 2.f);
    k_accum<<<ewg(NDZ), 256, 0, stream>>>(za, vs, DT / 3.f, NDZ);
    k_accum<<<ewg(NDZ), 256, 0, stream>>>(va, aF, DT / 3.f, NDZ);
    k_lincomb<<<ewg(NDZ), 256, 0, stream>>>(zs, zp, vs, DT / 2.f, NDZ);  // uses old vs (k2v)
    k_lincomb<<<ewg(NDZ), 256, 0, stream>>>(vs, vp, aF, DT / 2.f, NDZ);
    // stage 3 (k3v = vs, k3a = aF)
    force(zs, vs, t + DT / 2.f);
    k_accum<<<ewg(NDZ), 256, 0, stream>>>(za, vs, DT / 3.f, NDZ);
    k_accum<<<ewg(NDZ), 256, 0, stream>>>(va, aF, DT / 3.f, NDZ);
    k_lincomb<<<ewg(NDZ), 256, 0, stream>>>(zs, zp, vs, DT, NDZ);        // uses old vs (k3v)
    k_lincomb<<<ewg(NDZ), 256, 0, stream>>>(vs, vp, aF, DT, NDZ);
    // stage 4 (k4v = vs, k4a = aF)
    force(zs, vs, t + DT);
    k_accum<<<ewg(NDZ), 256, 0, stream>>>(za, vs, DT / 6.f, NDZ);
    k_accum<<<ewg(NDZ), 256, 0, stream>>>(va, aF, DT / 6.f, NDZ);
    // advance: swap state/accumulator pointers
    float* tz = zp; zp = za; za = tz;
    float* tv = vp; vp = va; va = tv;
  }

  k_concat<<<ewg(BATCH * 512), 256, 0, stream>>>(zp, vp, (float*)d_out, BATCH * 512);
}
